// CoarseScorer_87333864997261
// MI455X (gfx1250) — compile-verified, matrix operand order checked by reading, not run on hardware
//
#include <hip/hip_runtime.h>
#include <hip/hip_bf16.h>

#define N_TOK   16384
#define DIM     768
#define TOPK    50
#define KSTEPS  (DIM / 32)
#define NTILES_I (N_TOK / 16)
#define NTILES_C (DIM / 16)

typedef __bf16 bf16_t;
typedef bf16_t v16bf __attribute__((ext_vector_type(16)));
typedef float  v8f   __attribute__((ext_vector_type(8)));
typedef unsigned int u32x4 __attribute__((ext_vector_type(4)));

union FragBF {
    v16bf v;
    u32x4 q[2];
};

__device__ __forceinline__ unsigned short f32_to_bf16_rne(float f) {
    unsigned int u = __float_as_uint(f);
    u += 0x7fffu + ((u >> 16) & 1u);
    return (unsigned short)(u >> 16);
}
__device__ __forceinline__ float bf16_us_to_f32(unsigned short h) {
    return __uint_as_float(((unsigned int)h) << 16);
}

// Fragment load (global): lane<16 -> K offsets {0..7, 16..23}; lane>=16 ->
// {8..15, 24..31}. 'off' = (lane>>4)*8. Row data is contiguous bf16 (ld = DIM).
__device__ __forceinline__ v16bf load_frag(const unsigned short* __restrict__ base,
                                           int row, int kbase, int off) {
    const unsigned short* p = base + (size_t)row * DIM + kbase + off;
    FragBF f;
    f.q[0] = *(const u32x4*)(p);
    f.q[1] = *(const u32x4*)(p + 16);
    return f.v;
}

// Fragment load from an LDS-resident row (same element layout).
__device__ __forceinline__ v16bf load_frag_lds(const unsigned short* p) {
    FragBF f;
    f.q[0] = *(const u32x4*)(p);
    f.q[1] = *(const u32x4*)(p + 16);
    return f.v;
}

__device__ __forceinline__ v8f wmma_bf16(v16bf a, v16bf b, v8f c) {
    return __builtin_amdgcn_wmma_f32_16x16x32_bf16(false, a, false, b,
                                                   (short)0, c, false, false);
}

// ---------------------------------------------------------------------------
// Kernel 0: split fp32 E and W into bf16 hi/lo planes (hi + lo ~ fp32).
// ---------------------------------------------------------------------------
__global__ void coarse_prep_split(const float* __restrict__ E,
                                  const float* __restrict__ W,
                                  unsigned short* __restrict__ Ehi,
                                  unsigned short* __restrict__ Elo,
                                  unsigned short* __restrict__ Whi,
                                  unsigned short* __restrict__ Wlo) {
    const int nE = N_TOK * DIM;
    const int nW = DIM * DIM;
    int stride = gridDim.x * blockDim.x;
    for (int i = blockIdx.x * blockDim.x + threadIdx.x; i < nE + nW; i += stride) {
        float v;
        unsigned short *ph, *pl;
        int j;
        if (i < nE) { j = i; v = E[j]; ph = Ehi; pl = Elo; }
        else        { j = i - nE; v = W[j]; ph = Whi; pl = Wlo; }
        unsigned short h = f32_to_bf16_rne(v);
        unsigned short l = f32_to_bf16_rne(v - bf16_us_to_f32(h));
        ph[j] = h;
        pl[j] = l;
    }
}

// ---------------------------------------------------------------------------
// Kernel 1: proj = E @ W^T + b  via split-bf16 WMMA, re-split into Phi/Plo.
// One 16x16 output tile per wave; K = 768 in 24 steps of 32.
// B(k,c) = W[c,k] -> lane's B fragment is contiguous K of row c of W.
// ---------------------------------------------------------------------------
__global__ __launch_bounds__(256) void coarse_proj_wmma(
        const unsigned short* __restrict__ Ehi, const unsigned short* __restrict__ Elo,
        const unsigned short* __restrict__ Whi, const unsigned short* __restrict__ Wlo,
        const float* __restrict__ bias,
        unsigned short* __restrict__ Phi, unsigned short* __restrict__ Plo) {
    int wave = threadIdx.x >> 5;
    int lane = threadIdx.x & 31;
    int t = blockIdx.x * 8 + wave;
    int it = t / NTILES_C;
    int ct = t % NTILES_C;
    int ibase = it * 16, cbase = ct * 16;
    int nl = lane & 15, hiHalf = lane >> 4, off = hiHalf * 8;
    int rowA = ibase + nl;   // row of E
    int rowB = cbase + nl;   // row of W (B is W^T)
    v8f acc = {};
    for (int ks = 0; ks < KSTEPS; ++ks) {
        int kb = ks * 32;
        v16bf ah = load_frag(Ehi, rowA, kb, off);
        v16bf al = load_frag(Elo, rowA, kb, off);
        v16bf bh = load_frag(Whi, rowB, kb, off);
        v16bf bl = load_frag(Wlo, rowB, kb, off);
        acc = wmma_bf16(ah, bh, acc);
        acc = wmma_bf16(ah, bl, acc);
        acc = wmma_bf16(al, bh, acc);
    }
    int col = cbase + nl;
    float bv = bias[col];
    for (int r = 0; r < 8; ++r) {
        int row = ibase + r + hiHalf * 8;   // C/D layout: VGPR r = row r / r+8
        float v = acc[r] + bv;
        unsigned short h = f32_to_bf16_rne(v);
        unsigned short l = f32_to_bf16_rne(v - bf16_us_to_f32(h));
        size_t o = (size_t)row * DIM + col;
        Phi[o] = h;
        Plo[o] = l;
    }
}

// ---------------------------------------------------------------------------
// Kernel 2: fused triangular GEMM (proj @ E^T) + mask + char scores + top-50.
// Block ib owns rows [16*ib, 16*ib+16). The block's 16x768 proj stripe (hi+lo
// bf16) is staged into LDS once: the A fragments are identical for all 8
// waves and invariant over the whole jt loop, so this halves global-memory
// requests in the steady state (B comes from L2, A from ds_load_b128).
// Per-wave per-row top-50 candidate buffers in LDS; merged after barrier.
// ---------------------------------------------------------------------------
__global__ __launch_bounds__(256) void coarse_score_topk(
        const unsigned short* __restrict__ Phi, const unsigned short* __restrict__ Plo,
        const unsigned short* __restrict__ Ehi, const unsigned short* __restrict__ Elo,
        const float* __restrict__ charS,
        float* __restrict__ outScore, float* __restrict__ outIdx) {
    __shared__ unsigned short s_A[2][16][DIM];    // 48 KB: proj stripe hi/lo
    __shared__ float s_cand[8][16][TOPK];
    __shared__ int   s_cidx[8][16][TOPK];
    __shared__ float s_thr[8][16];
    __shared__ float s_char[16];
    const float NEG_INF = -__builtin_inff();

    int tid  = threadIdx.x;
    int wave = tid >> 5;
    int lane = tid & 31;
    int ib = blockIdx.x;
    int ibase = ib * 16;

    // Stage A stripe (hi/lo planes) into LDS, 16B at a time.
    for (int e = tid; e < 2 * 16 * (DIM / 8); e += 256) {
        int plane = e / (16 * (DIM / 8));
        int rm = e % (16 * (DIM / 8));
        int row = rm / (DIM / 8);
        int k8 = (rm % (DIM / 8)) * 8;
        const unsigned short* src =
            (plane ? Plo : Phi) + (size_t)(ibase + row) * DIM + k8;
        *(u32x4*)&s_A[plane][row][k8] = *(const u32x4*)src;
    }

    // Init: -inf candidates with ascending padding indices starting at row i,
    // so rows with < 50 valid entries reproduce jax's -inf/lowest-index padding.
    for (int e = tid; e < 8 * 16 * TOPK; e += 256) {
        int wv = e / (16 * TOPK);
        int rm = e % (16 * TOPK);
        int rr = rm / TOPK, sl = rm % TOPK;
        s_cand[wv][rr][sl] = NEG_INF;
        s_cidx[wv][rr][sl] = ibase + rr + wv * TOPK + sl;
    }
    if (tid < 128) s_thr[tid >> 4][tid & 15] = NEG_INF;
    if (tid < 16)  s_char[tid] = charS[ibase + tid];
    __syncthreads();

    int nl = lane & 15, hiHalf = lane >> 4, off = hiHalf * 8;

    for (int jt = wave; jt <= ib; jt += 8) {
        int jbase = jt * 16;
        int rowB = jbase + nl;   // B(k,j) = E[j,k]
        v8f acc = {};
        for (int ks = 0; ks < KSTEPS; ++ks) {
            int kb = ks * 32;
            v16bf ah = load_frag_lds(&s_A[0][nl][kb + off]);
            v16bf al = load_frag_lds(&s_A[1][nl][kb + off]);
            v16bf bh = load_frag(Ehi, rowB, kb, off);
            v16bf bl = load_frag(Elo, rowB, kb, off);
            acc = wmma_bf16(ah, bh, acc);
            acc = wmma_bf16(ah, bl, acc);
            acc = wmma_bf16(al, bh, acc);
        }
        float charJ = charS[jbase + nl];
        int colG = jbase + nl;
        for (int r = 0; r < 8; ++r) {
            int rowLocal = r + hiHalf * 8;
            int rowG = ibase + rowLocal;
            float v = acc[r] + s_char[rowLocal] + charJ;
            if (colG >= rowG) v = NEG_INF;              // pair mask: need j < i
            float thr = s_thr[wave][rowLocal];
            unsigned int mask = (unsigned int)__ballot(v > thr);
            while (mask) {
                int src = __ffs(mask) - 1;
                mask &= mask - 1;
                float cv = __shfl(v, src);
                int crow = r + ((src >= 16) ? 8 : 0);
                int ccol = jbase + (src & 15);
                float* cs = s_cand[wave][crow];
                int*   ci = s_cidx[wave][crow];
                // Wave-parallel argmin over 50 entries; ties -> HIGH slot so
                // low -inf padding slots survive with their ascending indices.
                float a = cs[lane];
                int   ap = lane;
                if (lane < TOPK - 32) {
                    float b2 = cs[lane + 32];
                    if (b2 <= a) { a = b2; ap = lane + 32; }
                }
                for (int o = 16; o; o >>= 1) {
                    float ov = __shfl_xor(a, o);
                    int   op = __shfl_xor(ap, o);
                    if (ov < a || (ov == a && op > ap)) { a = ov; ap = op; }
                }
                if (cv > a) {
                    if (lane == 0) {
                        cs[ap] = cv;
                        ci[ap] = ccol;
                        s_thr[wave][crow] = a;   // conservative threshold
                    }
                }
            }
        }
    }
    __syncthreads();

    // Merge: wave w handles rows {w, w+8}; 50 iterative argmax extractions
    // over the 8x50 candidates (tie-break -> lowest index, like lax.top_k).
    for (int rr = wave; rr < 16; rr += 8) {
        int rowG = ibase + rr;
        for (int sel = 0; sel < TOPK; ++sel) {
            float bv = NEG_INF;
            int bi = 0x7fffffff;
            int be = -1;
            for (int e = lane; e < 8 * TOPK; e += 32) {
                int wv = e / TOPK, sl = e % TOPK;
                float vv = s_cand[wv][rr][sl];
                int   ii = s_cidx[wv][rr][sl];
                if (vv > bv || (vv == bv && ii < bi)) { bv = vv; bi = ii; be = e; }
            }
            for (int o = 16; o; o >>= 1) {
                float ov = __shfl_xor(bv, o);
                int   oi = __shfl_xor(bi, o);
                int   oe = __shfl_xor(be, o);
                if (ov > bv || (ov == bv && oi < bi)) { bv = ov; bi = oi; be = oe; }
            }
            if (lane == 0) {
                outScore[(size_t)rowG * TOPK + sel] = bv;
                outIdx[(size_t)rowG * TOPK + sel]   = (float)bi;
                if (be >= 0) s_cand[be / TOPK][rr][be % TOPK] = NEG_INF;
            }
        }
    }
}

// ---------------------------------------------------------------------------
// Launch
// ---------------------------------------------------------------------------
extern "C" void kernel_launch(void* const* d_in, const int* in_sizes, int n_in,
                              void* d_out, int out_size, void* d_ws, size_t ws_size,
                              hipStream_t stream) {
    const float* E     = (const float*)d_in[0];   // word_embeddings [n, d]
    const float* charS = (const float*)d_in[1];   // character_scores [n]
    const float* W     = (const float*)d_in[2];   // W [d, d]
    const float* bias  = (const float*)d_in[3];   // b [d]
    // d_in[4] = topk (compile-time fixed at 50)

    char* ws = (char*)d_ws;
    const size_t nE = (size_t)N_TOK * DIM;
    const size_t nW = (size_t)DIM * DIM;
    unsigned short* Ehi = (unsigned short*)ws; ws += nE * 2;
    unsigned short* Elo = (unsigned short*)ws; ws += nE * 2;
    unsigned short* Whi = (unsigned short*)ws; ws += nW * 2;
    unsigned short* Wlo = (unsigned short*)ws; ws += nW * 2;
    unsigned short* Phi = (unsigned short*)ws; ws += nE * 2;
    unsigned short* Plo = (unsigned short*)ws; ws += nE * 2;

    float* outScore = (float*)d_out;
    float* outIdx   = outScore + (size_t)N_TOK * TOPK;

    hipLaunchKernelGGL(coarse_prep_split, dim3(2048), dim3(256), 0, stream,
                       E, W, Ehi, Elo, Whi, Wlo);
    hipLaunchKernelGGL(coarse_proj_wmma, dim3((NTILES_I * NTILES_C) / 8), dim3(256), 0, stream,
                       Ehi, Elo, Whi, Wlo, bias, Phi, Plo);
    hipLaunchKernelGGL(coarse_score_topk, dim3(NTILES_I), dim3(256), 0, stream,
                       Phi, Plo, Ehi, Elo, charS, outScore, outIdx);
}
